// AbstractODEDecoder_32031866093926
// MI455X (gfx1250) — matrix operand order repaired and checked by reading, not compiled, not run on hardware
//
#include <hip/hip_runtime.h>
#include <cstdint>
#include <cstddef>

#define B_  512
#define N_  200
#define ZD_ 256
#define LD_ 128
#define HD_ 512
#define T_  100
#define K_  101   // T+1 unique times

typedef __attribute__((ext_vector_type(16))) _Float16 v16h;
typedef __attribute__((ext_vector_type(8)))  float    v8f;

__device__ __forceinline__ v8f wmma16(v16h a, v16h b, v8f c) {
  // D = A(16x32 f16) * B(32x16 f16) + C(16x16 f32)
  return __builtin_amdgcn_wmma_f32_16x16x32_f16(
      /*neg_a=*/false, a, /*neg_b=*/false, b,
      /*c_mod=*/(short)0, c, /*reuse_a=*/false, /*reuse_b=*/false);
}

// Branch-free tanh: v_exp_f32 + v_rcp_f32, clamped so e^{2x} stays finite.
__device__ __forceinline__ float fast_tanh(float x) {
  float cx = fminf(fmaxf(x, -9.0f), 9.0f);
  float e2 = __expf(2.0f * cx);
  return (e2 - 1.0f) * __builtin_amdgcn_rcpf(e2 + 1.0f);
}

// A-operand load from an LDS row (row-major activations).
// ISA 16-bit A layout: lane L (M = L%16) holds K = kbase+off+0..7 and
// K = kbase+16+off+0..7 with off = (L<16 ? 0 : 8)  ->  two 16B ds_load_b128.
__device__ __forceinline__ v16h load_a_lds(const _Float16* p) {
  union U { uint4 q[2]; v16h v; } u;
  u.q[0] = *(const uint4*)(p);
  u.q[1] = *(const uint4*)(p + 16);
  return u.v;
}

// ---------------- weight packing into WMMA B-lane order ------------------
// B layout (16-bit, 32x16 tile): lane L holds K = off+0..7 and K = 16+off+0..7
// at column N = L%16, off = (L<16 ? 0 : 8) (matches A). Packed tile = 512 f16,
// lane-major: dst[tile*512 + lane*16 + e]  ->  one 32B contiguous load/lane.
__global__ void k_pack(const float* __restrict__ W, int rowOff, int Nsrc,
                       int Nt, int total, _Float16* __restrict__ dst) {
  int id = blockIdx.x * blockDim.x + threadIdx.x;
  if (id >= total) return;
  int tile   = id >> 9;
  int within = id & 511;
  int lane   = within >> 4;
  int e      = within & 15;
  int kt = tile / Nt, nt = tile - kt * Nt;
  int k = kt * 32 + ((lane & 16) ? 8 : 0) + ((e & 8) ? 16 : 0) + (e & 7);
  int n = nt * 16 + (lane & 15);
  dst[id] = (_Float16)W[(size_t)(k + rowOff) * Nsrc + n];
}

// rank-1 rows (t column of ode_W1, x column of dec_W1) + zero presence map
__global__ void k_misc(const float* __restrict__ odeW1,
                       const float* __restrict__ decW1,
                       float* __restrict__ w1t, float* __restrict__ d1x,
                       int* __restrict__ pres) {
  int id = threadIdx.x;
  if (id < HD_) { w1t[id] = odeW1[(size_t)ZD_ * HD_ + id]; d1x[id] = decW1[id]; }
  if (id < K_)  pres[id] = 0;
}

__global__ void k_presence(const float* __restrict__ x,
                           const float* __restrict__ it,
                           int* __restrict__ pres, int total) {
  int id = blockIdx.x * blockDim.x + threadIdx.x;
  if (id == 0) {
    int g0 = (int)(it[0] * (float)T_ + 0.5f);
    if (g0 >= 0 && g0 <= T_) pres[g0] = 1;
  }
  if (id < total) {
    int g = (int)(x[id] * (float)T_ + 0.5f);
    if (g >= 0 && g <= T_) pres[g] = 1;
  }
}

// sorted unique times (pad with 1.0, matching jnp.unique fill_value)
__global__ void k_times(const int* __restrict__ pres, float* __restrict__ times,
                        int* __restrict__ map) {
  if (threadIdx.x == 0 && blockIdx.x == 0) {
    int cnt = 0;
    for (int g = 0; g <= T_; ++g) {
      if (pres[g]) { map[g] = cnt; times[cnt] = (float)g / (float)T_; ++cnt; }
      else map[g] = 0;
    }
    for (int i = cnt; i < K_; ++i) times[i] = 1.0f;
  }
}

// ---------------- ODE scan: 16 batch rows per workgroup -------------------
// Batch rows are independent -> no inter-workgroup sync across the whole scan.
__global__ __launch_bounds__(512) void k_ode(
    const float* __restrict__ z,  const float* __restrict__ b1,
    const float* __restrict__ b2, const float* __restrict__ b3,
    const _Float16* __restrict__ w1p, const _Float16* __restrict__ w2p,
    const _Float16* __restrict__ w3p, const float* __restrict__ w1t,
    const float* __restrict__ times, float* __restrict__ pv) {
  __shared__ _Float16 a_h[16][LD_];      // evolving layer-1 input     4 KB
  __shared__ _Float16 h_h[16][HD_];      // activations (in-place L2) 16 KB
  __shared__ float    ks[5][16][LD_];    // RK stages (k6 -> slot 0)  40 KB

  const int tid  = threadIdx.x;
  const int lane = tid & 31;
  const int wave = tid >> 5;
  const int r0   = blockIdx.x * 16;
  const int mA   = lane & 15;
  const int offA = (lane & 16) ? 8 : 0;
  const int mrow = (lane & 16) ? 8 : 0;
  const int ncol = lane & 15;

  // each thread owns 4 evolving state elements (flattened m*128+c)
  float vr[4];
  int   midx[4], cidx[4];
#pragma unroll
  for (int j = 0; j < 4; ++j) {
    int idx = tid * 4 + j;
    int m = idx >> 7, c = idx & 127;
    midx[j] = m; cidx[j] = c;
    float zv = z[(size_t)(r0 + m) * ZD_ + c];
    vr[j] = zv;
    pv[((size_t)(r0 + m) * K_ + 0) * LD_ + c] = zv;   // v_all[0] = z[:, :LD]
  }

  // ---- pre-accumulate layer-1 contribution of the static half ----
  // z[:,128:256] never evolves: its GEMM partial lives in 16 resident VGPRs
  // per wave for the entire scan.  Stage layer-1 then only needs K=128.
  for (int i = tid; i < 16 * 128; i += 512) {
    int m = i >> 7, c = i & 127;
    h_h[m][c] = (_Float16)z[(size_t)(r0 + m) * ZD_ + (128 + c)];  // staging
  }
  __syncthreads();
  v8f c1i[2];
#pragma unroll
  for (int u = 0; u < 2; ++u) {
    int nt = wave * 2 + u;
    v8f acc = {0.f,0.f,0.f,0.f,0.f,0.f,0.f,0.f};
#pragma unroll
    for (int kt = 0; kt < 4; ++kt) {                 // K tiles 4..7 of W1
      v16h av = load_a_lds(&h_h[mA][kt * 32 + offA]);
      v16h bv = *(const v16h*)(w1p + ((size_t)((kt + 4) * 32 + nt) << 9) + (lane << 4));
      acc = wmma16(av, bv, acc);
    }
    c1i[u] = acc;
  }
  // staging area h_h is reused for activations after the next barrier

#pragma unroll 1
  for (int step = 0; step < K_ - 1; ++step) {
    float t0 = times[step], t1 = times[step + 1];
    float dt = t1 - t0;
#pragma unroll 1
    for (int s = 0; s < 6; ++s) {
      // compiler barrier: keep weight loads inside the stage (stream from L2,
      // never hoist-and-spill to scratch)
      asm volatile("" ::: "memory");
      // ---- stage input (only first LD coords evolve) ----
#pragma unroll
      for (int j = 0; j < 4; ++j) {
        int m = midx[j], c = cidx[j];
        float acc = vr[j];
        switch (s) {
          case 1: acc += dt * (0.2f * ks[0][m][c]); break;
          case 2: acc += dt * ((3.f/40.f)*ks[0][m][c] + (9.f/40.f)*ks[1][m][c]); break;
          case 3: acc += dt * ((44.f/45.f)*ks[0][m][c] - (56.f/15.f)*ks[1][m][c]
                             + (32.f/9.f)*ks[2][m][c]); break;
          case 4: acc += dt * ((19372.f/6561.f)*ks[0][m][c] - (25360.f/2187.f)*ks[1][m][c]
                             + (64448.f/6561.f)*ks[2][m][c] - (212.f/729.f)*ks[3][m][c]); break;
          case 5: acc += dt * ((9017.f/3168.f)*ks[0][m][c] - (355.f/33.f)*ks[1][m][c]
                             + (46732.f/5247.f)*ks[2][m][c] + (49.f/176.f)*ks[3][m][c]
                             - (5103.f/18656.f)*ks[4][m][c]); break;
          default: break;
        }
        a_h[m][c] = (_Float16)acc;
        if (s == 5) {
          // partial Butcher update (k6 term added after this stage); frees ks[0]
          vr[j] += dt * ((35.f/384.f)*ks[0][m][c] + (500.f/1113.f)*ks[2][m][c]
                       + (125.f/192.f)*ks[3][m][c] - (2187.f/6784.f)*ks[4][m][c]);
        }
      }
      __syncthreads();
      float cs = (s==0)?0.f:(s==1)?0.2f:(s==2)?0.3f:(s==3)?0.8f:(s==4)?(8.f/9.f):1.f;
      float ts = t0 + cs * dt;

      // ---- layer 1: [16x128]@[128x512] + static-half + (b1 + t*W1_t), tanh ----
#pragma unroll
      for (int u = 0; u < 2; ++u) {
        int nt = wave * 2 + u;
        v8f acc = c1i[u];
#pragma unroll
        for (int kt = 0; kt < 4; ++kt) {
          v16h av = load_a_lds(&a_h[mA][kt * 32 + offA]);
          v16h bv = *(const v16h*)(w1p + ((size_t)(kt * 32 + nt) << 9) + (lane << 4));
          acc = wmma16(av, bv, acc);
        }
        int n = nt * 16 + ncol;
        float bb = b1[n] + ts * w1t[n];
#pragma unroll
        for (int r = 0; r < 8; ++r) h_h[mrow + r][n] = (_Float16)fast_tanh(acc[r] + bb);
      }
      __syncthreads();

      // ---- layer 2: [16x512]@[512x512], tanh, in-place two-pass ----
      v8f acc2[2];
#pragma unroll
      for (int u = 0; u < 2; ++u) {
        int nt = wave * 2 + u;
        v8f acc = {0.f,0.f,0.f,0.f,0.f,0.f,0.f,0.f};
#pragma unroll 4
        for (int kt = 0; kt < 16; ++kt) {
          v16h av = load_a_lds(&h_h[mA][kt * 32 + offA]);
          v16h bv = *(const v16h*)(w2p + ((size_t)(kt * 32 + nt) << 9) + (lane << 4));
          acc = wmma16(av, bv, acc);
        }
        acc2[u] = acc;
      }
      __syncthreads();
#pragma unroll
      for (int u = 0; u < 2; ++u) {
        int n = (wave * 2 + u) * 16 + ncol;
        float bb = b2[n];
#pragma unroll
        for (int r = 0; r < 8; ++r) h_h[mrow + r][n] = (_Float16)fast_tanh(acc2[u][r] + bb);
      }
      __syncthreads();

      // ---- layer 3: [16x512]@[512x128] -> k_s ----
      if (wave < 8) {
        int nt = wave;
        v8f acc = {0.f,0.f,0.f,0.f,0.f,0.f,0.f,0.f};
#pragma unroll 4
        for (int kt = 0; kt < 16; ++kt) {
          v16h av = load_a_lds(&h_h[mA][kt * 32 + offA]);
          v16h bv = *(const v16h*)(w3p + ((size_t)(kt * 8 + nt) << 9) + (lane << 4));
          acc = wmma16(av, bv, acc);
        }
        int n = nt * 16 + ncol;
        float bb = b3[n];
        int slot = (s == 5) ? 0 : s;   // k6 reuses slot 0 (freed above)
#pragma unroll
        for (int r = 0; r < 8; ++r) ks[slot][mrow + r][n] = acc[r] + bb;
      }
      __syncthreads();
    }
    // finish update with k6 term; emit v_all[step+1][:, :LD]
#pragma unroll
    for (int j = 0; j < 4; ++j) {
      int m = midx[j], c = cidx[j];
      vr[j] += dt * (11.f/84.f) * ks[0][m][c];
      pv[((size_t)(r0 + m) * K_ + (step + 1)) * LD_ + c] = vr[j];
    }
    __syncthreads();
  }
}

// ---------------- decoder: 16 rows of (b,n) per workgroup -----------------
__global__ __launch_bounds__(512) void k_dec(
    const float* __restrict__ x,  const float* __restrict__ z,
    const float* __restrict__ pv, const int* __restrict__ map,
    const _Float16* __restrict__ d1p, const _Float16* __restrict__ d2p,
    const _Float16* __restrict__ d3p, const float* __restrict__ d1x,
    const float* __restrict__ db1, const float* __restrict__ db2,
    const float* __restrict__ db3, float* __restrict__ out) {
  __shared__ _Float16 a_h[16][ZD_];
  __shared__ _Float16 h_h[16][HD_];
  __shared__ float xv[16];
  __shared__ int   idx_s[16];

  const int tid  = threadIdx.x;
  const int lane = tid & 31;
  const int wave = tid >> 5;
  const int mA   = lane & 15;
  const int offA = (lane & 16) ? 8 : 0;
  const int mrow = (lane & 16) ? 8 : 0;
  const int ncol = lane & 15;
  const int r0   = blockIdx.x * 16;

  if (tid < 16) {
    int row = r0 + tid;
    float xx = x[row];
    xv[tid] = xx;
    idx_s[tid] = map[(int)(xx * (float)T_ + 0.5f)];
  }
  __syncthreads();
  // A = [latent(128) | z[:,128:256]] ; x handled as rank-1 after the GEMM
  for (int i = tid; i < 16 * ZD_; i += 512) {
    int m = i >> 8, c = i & 255;
    int row = r0 + m, b = row / N_;
    float val = (c < LD_) ? pv[((size_t)b * K_ + idx_s[m]) * LD_ + c]
                          : z[(size_t)b * ZD_ + c];
    a_h[m][c] = (_Float16)val;
  }
  __syncthreads();

  // layer 1 -> relu -> h_h
#pragma unroll
  for (int u = 0; u < 2; ++u) {
    int nt = wave * 2 + u;
    v8f acc = {0.f,0.f,0.f,0.f,0.f,0.f,0.f,0.f};
#pragma unroll 4
    for (int kt = 0; kt < 8; ++kt) {
      v16h av = load_a_lds(&a_h[mA][kt * 32 + offA]);
      v16h bv = *(const v16h*)(d1p + ((size_t)(kt * 32 + nt) << 9) + (lane << 4));
      acc = wmma16(av, bv, acc);
    }
    int n = nt * 16 + ncol;
#pragma unroll
    for (int r = 0; r < 8; ++r) {
      float v = acc[r] + db1[n] + xv[mrow + r] * d1x[n];
      h_h[mrow + r][n] = (_Float16)fmaxf(v, 0.f);
    }
  }
  __syncthreads();

  // layer 2 in-place two-pass
  v8f acc2[2];
#pragma unroll
  for (int u = 0; u < 2; ++u) {
    int nt = wave * 2 + u;
    v8f acc = {0.f,0.f,0.f,0.f,0.f,0.f,0.f,0.f};
#pragma unroll 4
    for (int kt = 0; kt < 16; ++kt) {
      v16h av = load_a_lds(&h_h[mA][kt * 32 + offA]);
      v16h bv = *(const v16h*)(d2p + ((size_t)(kt * 32 + nt) << 9) + (lane << 4));
      acc = wmma16(av, bv, acc);
    }
    acc2[u] = acc;
  }
  __syncthreads();
#pragma unroll
  for (int u = 0; u < 2; ++u) {
    int n = (wave * 2 + u) * 16 + ncol;
    float bb = db2[n];
#pragma unroll
    for (int r = 0; r < 8; ++r)
      h_h[mrow + r][n] = (_Float16)fmaxf(acc2[u][r] + bb, 0.f);
  }
  __syncthreads();

  // layer 3 -> relu -> out (f32)
#pragma unroll
  for (int u = 0; u < 2; ++u) {
    int nt = wave * 2 + u;
    v8f acc = {0.f,0.f,0.f,0.f,0.f,0.f,0.f,0.f};
#pragma unroll 4
    for (int kt = 0; kt < 16; ++kt) {
      v16h av = load_a_lds(&h_h[mA][kt * 32 + offA]);
      v16h bv = *(const v16h*)(d3p + ((size_t)(kt * 32 + nt) << 9) + (lane << 4));
      acc = wmma16(av, bv, acc);
    }
    int n = nt * 16 + ncol;
    float bb = db3[n];
#pragma unroll
    for (int r = 0; r < 8; ++r)
      out[(size_t)(r0 + mrow + r) * HD_ + n] = fmaxf(acc[r] + bb, 0.f);
  }
}

extern "C" void kernel_launch(void* const* d_in, const int* in_sizes, int n_in,
                              void* d_out, int out_size, void* d_ws, size_t ws_size,
                              hipStream_t stream) {
  (void)in_sizes; (void)n_in; (void)out_size; (void)ws_size;
  const float* x   = (const float*)d_in[0];
  const float* z   = (const float*)d_in[1];
  const float* it  = (const float*)d_in[2];
  const float* oW1 = (const float*)d_in[3];
  const float* ob1 = (const float*)d_in[4];
  const float* oW2 = (const float*)d_in[5];
  const float* ob2 = (const float*)d_in[6];
  const float* oW3 = (const float*)d_in[7];
  const float* ob3 = (const float*)d_in[8];
  const float* dW1 = (const float*)d_in[9];
  const float* db1 = (const float*)d_in[10];
  const float* dW2 = (const float*)d_in[11];
  const float* db2 = (const float*)d_in[12];
  const float* dW3 = (const float*)d_in[13];
  const float* db3 = (const float*)d_in[14];

  uint8_t* ws = (uint8_t*)d_ws;
  size_t off = 0;
  auto alloc = [&](size_t bytes) -> uint8_t* {
    uint8_t* p = ws + off;
    off += (bytes + 255) & ~(size_t)255;
    return p;
  };
  _Float16* w1p = (_Float16*)alloc((size_t)8  * 32 * 512 * 2);
  _Float16* w2p = (_Float16*)alloc((size_t)16 * 32 * 512 * 2);
  _Float16* w3p = (_Float16*)alloc((size_t)16 * 8  * 512 * 2);
  _Float16* d1p = (_Float16*)alloc((size_t)8  * 32 * 512 * 2);
  _Float16* d2p = (_Float16*)alloc((size_t)16 * 32 * 512 * 2);
  _Float16* d3p = (_Float16*)alloc((size_t)16 * 32 * 512 * 2);
  float* w1t    = (float*)alloc(HD_ * 4);
  float* d1x    = (float*)alloc(HD_ * 4);
  float* times  = (float*)alloc(K_ * 4);
  int*   map    = (int*)alloc((T_ + 1) * 4);
  int*   pres   = (int*)alloc((T_ + 1) * 4);
  float* pv     = (float*)alloc((size_t)B_ * K_ * LD_ * 4);

  k_misc<<<1, 512, 0, stream>>>(oW1, dW1, w1t, d1x, pres);
  k_presence<<<(B_ * N_ + 255) / 256, 256, 0, stream>>>(x, it, pres, B_ * N_);
  k_times<<<1, 32, 0, stream>>>(pres, times, map);
  { int tot = 8  * 32 * 512; k_pack<<<(tot + 255) / 256, 256, 0, stream>>>(oW1, 0, HD_, 32, tot, w1p); }
  { int tot = 16 * 32 * 512; k_pack<<<(tot + 255) / 256, 256, 0, stream>>>(oW2, 0, HD_, 32, tot, w2p); }
  { int tot = 16 * 8  * 512; k_pack<<<(tot + 255) / 256, 256, 0, stream>>>(oW3, 0, LD_, 8,  tot, w3p); }
  { int tot = 8  * 32 * 512; k_pack<<<(tot + 255) / 256, 256, 0, stream>>>(dW1, 1, HD_, 32, tot, d1p); }
  { int tot = 16 * 32 * 512; k_pack<<<(tot + 255) / 256, 256, 0, stream>>>(dW2, 0, HD_, 32, tot, d2p); }
  { int tot = 16 * 32 * 512; k_pack<<<(tot + 255) / 256, 256, 0, stream>>>(dW3, 0, HD_, 32, tot, d3p); }

  k_ode<<<B_ / 16, 512, 0, stream>>>(z, ob1, ob2, ob3, w1p, w2p, w3p, w1t, times, pv);
  k_dec<<<(B_ * N_) / 16, 512, 0, stream>>>(x, z, pv, map, d1p, d2p, d3p,
                                            d1x, db1, db2, db3, (float*)d_out);
}
